// Renderer_24893630447782
// MI455X (gfx1250) — matrix operand order repaired and checked by reading, not compiled, not run on hardware
//
#include <hip/hip_runtime.h>
#include <hip/hip_bf16.h>
#include <math.h>

#define NPTS   524288
#define NV     6890
#define NJ     24
#define AS     160          // activation row stride (halves)

typedef _Float16 __attribute__((ext_vector_type(16))) v16h;
typedef _Float16 __attribute__((ext_vector_type(8)))  v8h;
typedef float    __attribute__((ext_vector_type(8)))  v8f;

__constant__ int c_par[NJ] = {-1,0,0,0,1,2,3,4,5,6,7,8,9,9,9,12,13,14,16,17,18,19,20,21};

// ---------------------------------------------------------------------------
// Kernel 1: joints[j][k] = sum_v Jreg[j,v] * (v_template[v,k] + sum_b sd[v,k,b]*beta[b])
// 72 blocks (one per (j,k)), 256-thread reduction over V.
// ---------------------------------------------------------------------------
__global__ void k_joints(const float* __restrict__ Jreg, const float* __restrict__ vtemp,
                         const float* __restrict__ sdirs, const float* __restrict__ betas,
                         float* __restrict__ joints_out)
{
    __shared__ float red[256];
    const int jk = blockIdx.x;              // 0..71
    const int j = jk / 3, k = jk % 3;
    float b[10];
#pragma unroll
    for (int i = 0; i < 10; ++i) b[i] = betas[i];
    float acc = 0.f;
    for (int v = threadIdx.x; v < NV; v += 256) {
        float vs = vtemp[v*3 + k];
        const float* sd = sdirs + (v*3 + k) * 10;
#pragma unroll
        for (int i = 0; i < 10; ++i) vs += sd[i] * b[i];
        acc += Jreg[j*NV + v] * vs;
    }
    red[threadIdx.x] = acc;
    __syncthreads();
    for (int s = 128; s > 0; s >>= 1) {
        if (threadIdx.x < s) red[threadIdx.x] += red[threadIdx.x + s];
        __syncthreads();
    }
    if (threadIdx.x == 0) joints_out[jk] = red[0];
}

// ---------------------------------------------------------------------------
// Kernel 2: Rodrigues + kinematic chain + relative transforms A (24 x 4x4)
// ---------------------------------------------------------------------------
__global__ void k_prep(const float* __restrict__ poses, const float* __restrict__ joints_in,
                       float* __restrict__ A_out)
{
    __shared__ float Rm[NJ*9];
    __shared__ float tmS[NJ*16];
    __shared__ float chainS[NJ*16];
    __shared__ float jS[NJ*3];
    const int t = threadIdx.x;
    if (t < NJ*3) jS[t] = joints_in[t];
    if (t < NJ) {
        const int j = t;
        const float r0 = poses[3*j], r1 = poses[3*j+1], r2 = poses[3*j+2];
        const float a0 = r0 + 1e-8f, a1 = r1 + 1e-8f, a2 = r2 + 1e-8f;
        const float ang = sqrtf(a0*a0 + a1*a1 + a2*a2);
        const float inv = 1.f / ang;
        const float dx = r0*inv, dy = r1*inv, dz = r2*inv;
        const float cc = cosf(ang), ss = sinf(ang), omc = 1.f - cc;
        Rm[j*9+0] = 1.f + omc*(-(dy*dy + dz*dz));
        Rm[j*9+1] = -ss*dz + omc*dx*dy;
        Rm[j*9+2] =  ss*dy + omc*dx*dz;
        Rm[j*9+3] =  ss*dz + omc*dx*dy;
        Rm[j*9+4] = 1.f + omc*(-(dx*dx + dz*dz));
        Rm[j*9+5] = -ss*dx + omc*dy*dz;
        Rm[j*9+6] = -ss*dy + omc*dx*dz;
        Rm[j*9+7] =  ss*dx + omc*dy*dz;
        Rm[j*9+8] = 1.f + omc*(-(dx*dx + dy*dy));
    }
    __syncthreads();
    if (t < NJ) {
        const int j = t, p = c_par[j];
        float rx, ry, rz;
        if (j == 0) { rx = jS[0]; ry = jS[1]; rz = jS[2]; }
        else { rx = jS[j*3]-jS[p*3]; ry = jS[j*3+1]-jS[p*3+1]; rz = jS[j*3+2]-jS[p*3+2]; }
        tmS[j*16+ 0]=Rm[j*9+0]; tmS[j*16+ 1]=Rm[j*9+1]; tmS[j*16+ 2]=Rm[j*9+2]; tmS[j*16+ 3]=rx;
        tmS[j*16+ 4]=Rm[j*9+3]; tmS[j*16+ 5]=Rm[j*9+4]; tmS[j*16+ 6]=Rm[j*9+5]; tmS[j*16+ 7]=ry;
        tmS[j*16+ 8]=Rm[j*9+6]; tmS[j*16+ 9]=Rm[j*9+7]; tmS[j*16+10]=Rm[j*9+8]; tmS[j*16+11]=rz;
        tmS[j*16+12]=0.f; tmS[j*16+13]=0.f; tmS[j*16+14]=0.f; tmS[j*16+15]=1.f;
    }
    __syncthreads();
    if (t == 0) {   // sequential kinematic chain (parents[i] < i)
        for (int e = 0; e < 16; ++e) chainS[e] = tmS[e];
        for (int i = 1; i < NJ; ++i) {
            const int p = c_par[i];
            for (int a = 0; a < 4; ++a)
                for (int b = 0; b < 4; ++b) {
                    float s2 = 0.f;
                    for (int k = 0; k < 4; ++k) s2 += chainS[p*16 + a*4 + k] * tmS[i*16 + k*4 + b];
                    chainS[i*16 + a*4 + b] = s2;
                }
        }
    }
    __syncthreads();
    if (t < NJ) {
        const int j = t;
        float o[16];
#pragma unroll
        for (int e = 0; e < 16; ++e) o[e] = chainS[j*16 + e];
#pragma unroll
        for (int a = 0; a < 3; ++a) {
            const float relj = o[a*4+0]*jS[j*3+0] + o[a*4+1]*jS[j*3+1] + o[a*4+2]*jS[j*3+2];
            o[a*4+3] -= relj;
        }
#pragma unroll
        for (int e = 0; e < 16; ++e) A_out[j*16 + e] = o[e];
    }
}

// ---------------------------------------------------------------------------
// Kernel 3: weight transpose + f16 convert: dst[n*Kpad+k] = (f16) src[k*N+n], zero-pad k>=K
// ---------------------------------------------------------------------------
__global__ void k_wt(const float* __restrict__ src, _Float16* __restrict__ dst,
                     int K, int N, int Kpad)
{
    const int i = blockIdx.x * blockDim.x + threadIdx.x;
    if (i >= N * Kpad) return;
    const int n = i / Kpad, k = i - n * Kpad;
    const float v = (k < K) ? src[k * N + n] : 0.f;
    dst[i] = (_Float16)v;
}

// ---------------------------------------------------------------------------
// Fused skinning + tri-plane sampling + WMMA MLP
// ---------------------------------------------------------------------------
struct MlpParams {
    const float* coords; const float* viewdirs; const float* Rg; const float* Th;
    const float* bweights; const float* pf; const float* Aws;
    const _Float16* w0t; const _Float16* w1t; const _Float16* w2t;
    const _Float16* wft; const _Float16* wvt;
    const float* b0; const float* b1; const float* b2; const float* bfb;
    const float* wa; const float* ba; const float* bv; const float* wr; const float* br;
    float* out;
};

__device__ inline float fetch_tex(const float* __restrict__ base, int x, int y)
{
    return ((unsigned)x < 256u && (unsigned)y < 256u) ? base[(y << 8) + x] : 0.f;
}

__device__ inline v16h cat16(v8h lo, v8h hi)
{
    v16h r;
#pragma unroll
    for (int i = 0; i < 8; ++i) { r[i] = lo[i]; r[i+8] = hi[i]; }
    return r;
}

// A fragment (16x32 f16): lane holds row lane%16; halves 0-7 => K = kk+base0+0..7,
// halves 8-15 => K = kk+base0+16..23, base0 = (lane>=16)?8:0.  Two 16B LDS loads.
__device__ inline v16h load_afrag(const _Float16* act, int lane, int kk)
{
    const int row = lane & 15;
    const int b0  = (lane >> 4) << 3;
    const _Float16* p = act + row*AS + kk + b0;
    return cat16(*(const v8h*)p, *(const v8h*)(p + 16));
}

// B fragment (32x16 f16): lane holds col lane%16; halves 0-7 => K = kk+base0+0..7,
// halves 8-15 => K = kk+base0+16..23, base0 = (lane>=16)?8:0 (weights transposed [N][Kpad]).
__device__ inline v16h load_bfrag(const _Float16* __restrict__ wt, int Kpad, int lane, int nt, int kk)
{
    const int n = nt*16 + (lane & 15);
    const _Float16* p = wt + n*Kpad + kk + ((lane >> 4) << 3);
    return cat16(*(const v8h*)p, *(const v8h*)(p + 16));
}

// kk-outer / nt-inner: one A-fragment LDS load feeds NT back-to-back WMMAs;
// all NT C-tiles accumulate in registers (NT<=8 -> 64 VGPRs of accumulators).
template<int KPAD, int NT, bool RELU>
__device__ inline void gemm_layer(const _Float16* src, const _Float16* __restrict__ wt,
                                  const float* __restrict__ bias, _Float16* dst,
                                  int dstCol, int lane)
{
    v8f c[NT];
#pragma unroll
    for (int nt = 0; nt < NT; ++nt) {
        v8f z = {0.f,0.f,0.f,0.f,0.f,0.f,0.f,0.f};
        c[nt] = z;
    }
#pragma unroll
    for (int kk = 0; kk < KPAD; kk += 32) {
        const v16h a = load_afrag(src, lane, kk);
#pragma unroll
        for (int nt = 0; nt < NT; ++nt) {
            const v16h b = load_bfrag(wt, KPAD, lane, nt, kk);
            c[nt] = __builtin_amdgcn_wmma_f32_16x16x32_f16(false, a, false, b, (short)0, c[nt], false, false);
        }
    }
    const int rb = (lane >> 4) << 3;         // C layout: vgpr r -> row r + rb
#pragma unroll
    for (int nt = 0; nt < NT; ++nt) {
        const int col = nt*16 + (lane & 15);
        const float bb = bias[col];
#pragma unroll
        for (int r = 0; r < 8; ++r) {
            float v = c[nt][r] + bb;
            if (RELU) v = fmaxf(v, 0.f);
            dst[(rb + r)*AS + dstCol + col] = (_Float16)v;
        }
    }
}

__global__ __launch_bounds__(128) void k_mlp(MlpParams P)
{
    __shared__ __align__(16) _Float16 actS[4][2][16*AS];   // per-wave ping/pong act buffers
    __shared__ float Asm[NJ*16];

    const int t    = threadIdx.x;
    const int wave = t >> 5;
    const int lane = t & 31;
    const int row  = lane & 15;
    const int half = lane >> 4;
    const int tile = blockIdx.x * 4 + wave;
    const int m    = tile * 16 + row;

    for (int i = t; i < NJ*16; i += 128) Asm[i] = P.Aws[i];

    _Float16* actA = &actS[wave][0][0];
    _Float16* actB = &actS[wave][1][0];

    { // zero actA (each lane clears 80 halves of its row)
        v8h z = {(_Float16)0.f,(_Float16)0.f,(_Float16)0.f,(_Float16)0.f,
                 (_Float16)0.f,(_Float16)0.f,(_Float16)0.f,(_Float16)0.f};
        _Float16* zp = actA + row*AS + half*80;
#pragma unroll
        for (int i = 0; i < 10; ++i) *(v8h*)(zp + i*8) = z;
    }
    __syncthreads();

    // ---- LBS skinning: pts = (coord - Th) @ Rg; T = sum_j bw[j]*A[j]; deform ----
    float Rg[9];
#pragma unroll
    for (int i = 0; i < 9; ++i) Rg[i] = P.Rg[i];
    const float px = P.coords[m*3+0] - P.Th[0];
    const float py = P.coords[m*3+1] - P.Th[1];
    const float pz = P.coords[m*3+2] - P.Th[2];
    const float q0 = px*Rg[0] + py*Rg[3] + pz*Rg[6];
    const float q1 = px*Rg[1] + py*Rg[4] + pz*Rg[7];
    const float q2 = px*Rg[2] + py*Rg[5] + pz*Rg[8];

    float T[12];
#pragma unroll
    for (int e = 0; e < 12; ++e) T[e] = 0.f;
    const float* bw = P.bweights + m*NJ;
    for (int j = 0; j < NJ; ++j) {
        const float w = bw[j];
        const float* Aj = Asm + j*16;
#pragma unroll
        for (int e = 0; e < 12; ++e) T[e] += w * Aj[e];
    }
    const float inv_box = 1.f / 1.3f;
    const float nx = (T[0]*q0 + T[1]*q1 + T[2] *q2 + T[3] ) * inv_box;
    const float ny = (T[4]*q0 + T[5]*q1 + T[6] *q2 + T[7] ) * inv_box;
    const float nz = (T[8]*q0 + T[9]*q1 + T[10]*q2 + T[11]) * inv_box;

    // ---- tri-plane bilinear sampling: this lane covers 9 of 18 channels ----
    for (int ci = 0; ci < 9; ++ci) {
        const int c = half*9 + ci;
        const int g = c / 6;                       // 0: none, 1: u+=1/H, 2: v+=1/H
        float acc = 0.f;
#pragma unroll
        for (int p = 0; p < 3; ++p) {
            float u = (p == 2) ? nz : nx;          // p0:(nx,ny) p1:(nx,nz) p2:(nz,ny)
            float v = (p == 1) ? nz : ny;
            if (g == 1) u += (1.f/256.f);
            if (g == 2) v += (1.f/256.f);
            const float xf = (u + 1.f)*128.f - 0.5f;
            const float yf = (v + 1.f)*128.f - 0.5f;
            const float x0 = floorf(xf), y0 = floorf(yf);
            const float wx = xf - x0,   wy = yf - y0;
            const int xi = (int)x0, yi = (int)y0;
            const float* base = P.pf + ((p*18 + c) << 16);
            const float v00 = fetch_tex(base, xi,   yi);
            const float v01 = fetch_tex(base, xi+1, yi);
            const float v10 = fetch_tex(base, xi,   yi+1);
            const float v11 = fetch_tex(base, xi+1, yi+1);
            acc += v00*(1.f-wx)*(1.f-wy) + v01*wx*(1.f-wy)
                 + v10*(1.f-wx)*wy       + v11*wx*wy;
        }
        actA[row*AS + c] = (_Float16)(acc * (1.f/3.f));
    }

    // ---- posenc(viewdir): 27 values, split 14/13 across the two half-waves ----
    const float d0 = P.viewdirs[m*3+0];
    const float d1 = P.viewdirs[m*3+1];
    const float d2 = P.viewdirs[m*3+2];
    float vdv[14];
#pragma unroll
    for (int i = 0; i < 14; ++i) {
        const int e = half*14 + i;
        float val = 0.f;
        if (e < 3) {
            val = (e == 0) ? d0 : ((e == 1) ? d1 : d2);
        } else if (e < 27) {
            const int ee = e - 3;
            const int comp = ee >> 3;
            const int inner = ee & 7;
            const float dd = (comp == 0) ? d0 : ((comp == 1) ? d1 : d2);
            const float ang = dd * (float)(1 << (inner & 3));
            val = (inner < 4) ? sinf(ang) : cosf(ang);
        }
        vdv[i] = val;
    }
    __syncthreads();

    // ---- MLP via WMMA (activation cols: actA = [x(0..17)|h1(18..145)|0], actB = [h|0]) ----
    gemm_layer<32,  8, true >(actA, P.w0t, P.b0,  actB, 0,  lane);  // h0 = relu(x@W0+b0)
    __syncthreads();
    gemm_layer<128, 8, true >(actB, P.w1t, P.b1,  actA, 18, lane);  // h1 -> actA cols 18..145
    __syncthreads();
    gemm_layer<160, 8, true >(actA, P.w2t, P.b2,  actB, 0,  lane);  // h2 = relu([x,h1]@W2+b2)
    __syncthreads();

    float alpha = 0.f;
    if (half == 0) {                     // alpha = h2 @ Wa + ba   (lane = point row)
        float s = P.ba[0];
        const _Float16* hrow = actB + row*AS;
#pragma unroll
        for (int cb = 0; cb < 128; cb += 8) {
            const v8h hv = *(const v8h*)(hrow + cb);
#pragma unroll
            for (int i = 0; i < 8; ++i) s += (float)hv[i] * P.wa[cb + i];
        }
        alpha = s;
    }

    gemm_layer<128, 8, false>(actB, P.wft, P.bfb, actA, 0, lane);   // feat -> actA 0..127
#pragma unroll
    for (int i = 0; i < 14; ++i) {       // vd -> actA cols 128..154 (155..159 stay zero)
        const int e = half*14 + i;
        if (e < 27) actA[row*AS + 128 + e] = (_Float16)vdv[i];
    }
    __syncthreads();
    gemm_layer<160, 4, true >(actA, P.wvt, P.bv,  actB, 0, lane);   // hv = relu([feat,vd]@Wv+bv)
    __syncthreads();

    if (half == 0) {                     // rgb = hv @ Wr + br; write [rgb, alpha]
        float r0 = P.br[0], r1 = P.br[1], r2 = P.br[2];
        const _Float16* hrow = actB + row*AS;
#pragma unroll
        for (int cb = 0; cb < 64; cb += 8) {
            const v8h hv = *(const v8h*)(hrow + cb);
#pragma unroll
            for (int i = 0; i < 8; ++i) {
                const float h = (float)hv[i];
                r0 += h * P.wr[(cb + i)*3 + 0];
                r1 += h * P.wr[(cb + i)*3 + 1];
                r2 += h * P.wr[(cb + i)*3 + 2];
            }
        }
        float* o = P.out + m*4;
        o[0] = r0; o[1] = r1; o[2] = r2; o[3] = alpha;
    }
}

// ---------------------------------------------------------------------------
extern "C" void kernel_launch(void* const* d_in, const int* in_sizes, int n_in,
                              void* d_out, int out_size, void* d_ws, size_t ws_size,
                              hipStream_t stream)
{
    (void)out_size; (void)ws_size;
    const float* coords   = (const float*)d_in[0];
    const float* viewdirs = (const float*)d_in[1];
    const float* poses    = (const float*)d_in[2];
    const float* betas    = (const float*)d_in[3];
    const float* Rg       = (const float*)d_in[4];
    const float* Th       = (const float*)d_in[5];
    const float* bweights = (const float*)d_in[6];
    const float* pf       = (const float*)d_in[7];
    const float* vtemp    = (const float*)d_in[8];
    const float* sdirs    = (const float*)d_in[9];
    const float* Jreg     = (const float*)d_in[10];

    // mlp dict flatten order: default insertion (W0,b0,W1,b1,...); detect alphabetical.
    int iW0=11, ib0=12, iW1=13, ib1=14, iW2=15, ib2=16, iWf=17, ibf=18,
        iWa=19, iba=20, iWv=21, ibv=22, iWr=23, ibr=24;
    if (n_in >= 25 && in_sizes[12] == 128*128) {   // alphabetical: W0 W1 W2 Wa Wf Wr Wv b0..bv
        iW0=11; iW1=12; iW2=13; iWa=14; iWf=15; iWr=16; iWv=17;
        ib0=18; ib1=19; ib2=20; iba=21; ibf=22; ibr=23; ibv=24;
    }

    // workspace layout
    float*    Aws    = (float*)d_ws;                         // 24*16 f32
    float*    joints = (float*)((char*)d_ws + 1536);         // 72 f32
    _Float16* wt     = (_Float16*)((char*)d_ws + 2048);      // transposed f16 weights
    _Float16* w0t = wt;                  // 128*32
    _Float16* w1t = w0t + 128*32;        // 128*128
    _Float16* w2t = w1t + 128*128;       // 128*160
    _Float16* wft = w2t + 128*160;       // 128*128
    _Float16* wvt = wft + 128*128;       // 64*160

    k_joints<<<72, 256, 0, stream>>>(Jreg, vtemp, sdirs, betas, joints);
    k_prep  <<<1, 128, 0, stream>>>(poses, joints, Aws);
    k_wt<<<(128*32  + 255)/256, 256, 0, stream>>>((const float*)d_in[iW0], w0t,  18, 128,  32);
    k_wt<<<(128*128 + 255)/256, 256, 0, stream>>>((const float*)d_in[iW1], w1t, 128, 128, 128);
    k_wt<<<(128*160 + 255)/256, 256, 0, stream>>>((const float*)d_in[iW2], w2t, 146, 128, 160);
    k_wt<<<(128*128 + 255)/256, 256, 0, stream>>>((const float*)d_in[iWf], wft, 128, 128, 128);
    k_wt<<<(64*160  + 255)/256, 256, 0, stream>>>((const float*)d_in[iWv], wvt, 155,  64, 160);

    MlpParams P;
    P.coords = coords; P.viewdirs = viewdirs; P.Rg = Rg; P.Th = Th;
    P.bweights = bweights; P.pf = pf; P.Aws = Aws;
    P.w0t = w0t; P.w1t = w1t; P.w2t = w2t; P.wft = wft; P.wvt = wvt;
    P.b0  = (const float*)d_in[ib0];
    P.b1  = (const float*)d_in[ib1];
    P.b2  = (const float*)d_in[ib2];
    P.bfb = (const float*)d_in[ibf];
    P.wa  = (const float*)d_in[iWa];
    P.ba  = (const float*)d_in[iba];
    P.bv  = (const float*)d_in[ibv];
    P.wr  = (const float*)d_in[iWr];
    P.br  = (const float*)d_in[ibr];
    P.out = (float*)d_out;

    k_mlp<<<NPTS/64, 128, 0, stream>>>(P);   // 4 waves/block, one 16-pt tile per wave
}
